// PrimaryCaps_53687091200484
// MI455X (gfx1250) — compile-verified, matrix operand order checked by reading, not run on hardware
//
#include <hip/hip_runtime.h>

// ---- CDNA5 (gfx1250) wave32 WMMA implicit-GEMM conv + fused capsule squash ----
// Conv: x(64,256,40,40) * W(256,256,9,9) stride2 VALID -> (64,256,16,16) + b,
// then capsule squash over groups of 8 consecutive ow positions.
//
// GEMM: M=256 (out ch), N=16384 (b*256 + oh*16 + ow), K=20736.
// K reordered as k' = r*256 + c (r = kh*9+kw): every 32-wide K chunk has a
// single (kh,kw) and consecutive c -> one /9 decode per iteration and pure
// immediate-offset gathers. Weights are pre-permuted + downconverted to bf16
// in workspace so A fragments load straight global->VGPR in the ISA layout.

typedef __bf16 v16bf __attribute__((ext_vector_type(16)));
typedef __bf16 v8bf  __attribute__((ext_vector_type(8)));
typedef float  v8f   __attribute__((ext_vector_type(8)));

#define K_TOTAL 20736   // 81 * 256
#define KC      32      // K per bf16 WMMA step
#define MT      128     // block M tile
#define NT      128     // block N tile
#define LDB     40      // padded LDS row stride (elements): conflict-free b128

// One-time permute+downconvert: Wr[m][r*256+c] (bf16) = W[m][c*81+r] (f32).
// grid = (81, 256), block = 256: r = blockIdx.x, m = blockIdx.y, c = tid.
__global__ void reorder_W_kernel(const float* __restrict__ W,
                                 __bf16* __restrict__ Wr) {
    const int r = blockIdx.x;       // 0..80
    const int m = blockIdx.y;       // 0..255
    const int c = threadIdx.x;      // 0..255
    Wr[(size_t)m * K_TOTAL + r * 256 + c] =
        (__bf16)W[(size_t)m * K_TOTAL + c * 81 + r];
}

template <bool USE_WS>
__global__ __launch_bounds__(256, 2)
void caps_conv_wmma(const float* __restrict__ x,
                    const float* __restrict__ Wt,   // original f32 OIHW
                    const __bf16* __restrict__ Wr,  // reordered bf16 (USE_WS)
                    const float* __restrict__ bias,
                    float* __restrict__ out)
{
    // double-buffered B tile: 2 x 128 n x 32 k (bf16), padded stride
    __shared__ __align__(16) __bf16 Blds[2][NT * LDB];

    const int tid = threadIdx.x;
    const int l16 = tid & 15;
    const int hi  = (tid >> 4) & 1;        // half-wave select
    const int wv  = tid >> 5;              // wave id 0..7
    const int wm  = (wv >> 1) * 32;        // wave M offset: 0,32,64,96
    const int wn  = (wv & 1) * 64;         // wave N offset: 0,64

    const int m0 = blockIdx.x * MT;        // 0 or 128
    const int n0 = blockIdx.y * NT;        // spatial tile base

    // ---- B loader role: 128 n x 32 k; lanes of a wave take consecutive n ----
    const int bn  = tid & 127;             // 0..127
    const int bk0 = (tid >> 7) * 8;        // 0 or 8 (second chunk at +16)
    const int n_g = n0 + bn;               // n = b*256 + oh*16 + ow
    const int bb  = n_g >> 8;
    const int ss  = n_g & 255;
    const int oh  = ss >> 4;
    const int ow  = ss & 15;
    // x flat = b*409600 + c*1600 + (2*oh+kh)*40 + 2*ow+kw
    const int base_n = bb * 409600 + oh * 80 + ow * 2;

    // per-lane k-chunk split per ISA 16-bit fragment layout
    const int bk = hi ? 8 : 0;             // lanes<16: {0..7,16..23}; else {8..15,24..31}

    v8f acc[2][4] = {};

    int buf = 0;
    for (int kk = 0; kk < K_TOTAL; kk += KC, buf ^= 1) {
        // ---- decode chunk (one /9 per thread per iteration) ----
        const unsigned r  = (unsigned)kk >> 8;      // kernel tap, constant in chunk
        const unsigned kh = r / 9u;
        const unsigned kw = r - kh * 9u;
        const int      c0 = (kk & 255) + bk0;

        // ---- B gather: 16 immediate-offset loads, stride 1600 floats ----
        const float* xb = x + (base_n + (int)(kh * 40u + kw) + c0 * 1600);
        float breg[16];
        #pragma unroll
        for (int j = 0; j < 8; ++j) {
            breg[j]     = xb[j * 1600];
            breg[8 + j] = xb[(16 + j) * 1600];
        }

        // ---- A fragments: direct global -> VGPR in ISA layout ----
        union Frag { v16bf v; v8bf h[2]; __bf16 e[16]; };
        Frag afr[2];
        #pragma unroll
        for (int mi = 0; mi < 2; ++mi) {
            const int row = m0 + wm + mi * 16 + l16;
            if (USE_WS) {
                const __bf16* gp = Wr + (size_t)row * K_TOTAL + kk + bk;
                afr[mi].h[0] = *(const v8bf*)gp;
                afr[mi].h[1] = *(const v8bf*)(gp + 16);
                if (kk + KC < K_TOTAL)
                    __builtin_prefetch(gp + KC, 0, 1);   // global_prefetch_b8
            } else {
                // fallback: gather from original f32 W, stride 81 floats
                const float* gp = Wt + (size_t)row * K_TOTAL
                                + (size_t)((kk & 255) + bk) * 81 + r;
                #pragma unroll
                for (int e = 0; e < 8; ++e) {
                    afr[mi].e[e]     = (__bf16)gp[e * 81];
                    afr[mi].e[8 + e] = (__bf16)gp[(16 + e) * 81];
                }
            }
        }

        // ---- convert + stage B tile into current buffer (two b128 stores) ----
        {
            union { v8bf v; __bf16 e[8]; } p0, p1;
            #pragma unroll
            for (int j = 0; j < 8; ++j) {
                p0.e[j] = (__bf16)breg[j];
                p1.e[j] = (__bf16)breg[8 + j];
            }
            *(v8bf*)&Blds[buf][bn * LDB + bk0]      = p0.v;
            *(v8bf*)&Blds[buf][bn * LDB + bk0 + 16] = p1.v;
        }

        __syncthreads();   // single barrier: stores visible, prior buffer free

        // ---- B fragments from LDS + 8 WMMAs ----
        Frag bfr[4];
        #pragma unroll
        for (int ni = 0; ni < 4; ++ni) {
            const int col = wn + ni * 16 + l16;
            bfr[ni].h[0] = *(const v8bf*)&Blds[buf][col * LDB + bk];
            bfr[ni].h[1] = *(const v8bf*)&Blds[buf][col * LDB + bk + 16];
        }

        #pragma unroll
        for (int mi = 0; mi < 2; ++mi)
            #pragma unroll
            for (int ni = 0; ni < 4; ++ni)
                acc[mi][ni] = __builtin_amdgcn_wmma_f32_16x16x32_bf16(
                    /*neg_a=*/false, afr[mi].v,
                    /*neg_b=*/false, bfr[ni].v,
                    /*c_mod=*/(short)0, acc[mi][ni],
                    /*reuse_a=*/false, /*reuse_b=*/false);
    }

    // ---- epilogue: bias + capsule squash (8 consecutive ow = 8 lanes) ----
    // C/D layout: VGPR r holds M = r + 8*hi, lane l16 holds N = l16.
    #pragma unroll
    for (int mi = 0; mi < 2; ++mi) {
        const int chbase = m0 + wm + mi * 16 + hi * 8;
        #pragma unroll
        for (int ni = 0; ni < 4; ++ni) {
            const int ng = n0 + wn + ni * 16 + l16;
            const int b  = ng >> 8;
            const int s  = ng & 255;
            const int obase = b * 65536 + s;
            #pragma unroll
            for (int r = 0; r < 8; ++r) {
                const int ch = chbase + r;
                float v  = acc[mi][ni][r] + bias[ch];
                float sq = v * v;
                sq += __shfl_xor(sq, 1, 32);
                sq += __shfl_xor(sq, 2, 32);
                sq += __shfl_xor(sq, 4, 32);
                const float scale = sq / (1.0f + sq) * rsqrtf(sq + 1e-8f);
                out[obase + ch * 256] = scale * v;
            }
        }
    }
}

extern "C" void kernel_launch(void* const* d_in, const int* in_sizes, int n_in,
                              void* d_out, int out_size, void* d_ws, size_t ws_size,
                              hipStream_t stream) {
    (void)in_sizes; (void)n_in; (void)out_size;
    const float* x  = (const float*)d_in[0];
    const float* Wt = (const float*)d_in[1];
    const float* b  = (const float*)d_in[2];
    float* out = (float*)d_out;

    dim3 grid(2, 128, 1);     // M: 2x128, N: 128x128
    dim3 block(256, 1, 1);

    const size_t ws_need = (size_t)256 * K_TOTAL * sizeof(__bf16);  // 10.6 MB
    if (ws_size >= ws_need) {
        __bf16* Wr = (__bf16*)d_ws;
        reorder_W_kernel<<<dim3(81, 256, 1), dim3(256, 1, 1), 0, stream>>>(Wt, Wr);
        caps_conv_wmma<true><<<grid, block, 0, stream>>>(x, Wt, Wr, b, out);
    } else {
        caps_conv_wmma<false><<<grid, block, 0, stream>>>(x, Wt, nullptr, b, out);
    }
}
